// GraphLaplacianTransformerBackbone_53128745452156
// MI455X (gfx1250) — matrix operand order repaired and checked by reading, not compiled, not run on hardware
//
#include <hip/hip_runtime.h>
#include <math.h>

// ---- problem constants (from reference) ----
#define NN 50000
#define EE 500000
#define DD 128
#define HH 8
#define HD 16
#define SCALE 0.25f   // HD^-0.5

#define XSTR 136          // LDS row stride in f16 (16B row alignment + bank spread)
#define TILE_B (128 * XSTR)   // f16 elements per 128x136 LDS tile
#define TILE_BYTES (TILE_B * 2)

typedef _Float16 f16;
typedef __attribute__((ext_vector_type(16))) _Float16 v16h;
typedef __attribute__((ext_vector_type(8)))  _Float16 v8h;
typedef __attribute__((ext_vector_type(4)))  _Float16 v4h;
typedef __attribute__((ext_vector_type(8)))  float    v8f;
typedef __attribute__((ext_vector_type(4)))  unsigned int u32x4;
typedef __attribute__((ext_vector_type(8)))  int          i32x8;
typedef __attribute__((ext_vector_type(4)))  int          i32x4;

// ---------------- WMMA helpers (CDNA5 16x16x32 f16 -> f32) ----------------
__device__ __forceinline__ v16h cat16(v8h lo, v8h hi) {
    v16h r;
#pragma unroll
    for (int i = 0; i < 8; ++i) { r[i] = lo[i]; r[i + 8] = hi[i]; }
    return r;
}

// A-matrix 16x32 f16 fragment per ISA layout:
// lanes 0-15: M=lane,    K = {0..7} in v0-3, {16..23} in v4-7  (khalf=0)
// lanes16-31: M=lane-16, K = 8+{0..7} / 8+{16..23}             (khalf=8)
__device__ __forceinline__ v16h load_afrag(const f16* xs, int row, int kk, int khalf) {
    const f16* p = xs + row * XSTR + kk * 32 + khalf;
    v8h lo = *(const v8h*)p;
    v8h hi = *(const v8h*)(p + 16);
    return cat16(lo, hi);
}

// B-matrix 32x16 f16 fragment: lanes 0-15 hold K={0..7,16..23}, lanes 16-31 K={8..15,24..31}.
// Weight tile in LDS is [n][k] with row stride XSTR, so 8 K's are contiguous.
__device__ __forceinline__ v16h load_bfrag(const f16* wT, int n, int kbase) {
    const f16* p = wT + n * XSTR + kbase - ((kbase & 31) >> 1);
    v8h lo = *(const v8h*)p;
    v8h hi = *(const v8h*)(p + 16);
    return cat16(lo, hi);
}

__device__ __forceinline__ v8f wmma_f16(v16h a, v16h b, v8f c) {
    return __builtin_amdgcn_wmma_f32_16x16x32_f16(false, a, false, b, (short)0, c, false, false);
}

// ---------------- TDM: DMA one 128x128 f16 weight tile global->LDS ----------------
// 1-D tile of 16384 2-byte elements; pad_enable inserts 4 DWORDs (16B) of pad every
// 64 DWORDs (256B row) so the LDS image lands with row stride 272B == XSTR f16.
__device__ __forceinline__ unsigned int lds_off_u32(const void* p) {
    typedef __attribute__((address_space(3))) const char lds_char;
    return (unsigned int)(__UINTPTR_TYPE__)(lds_char*)p;
}

__device__ __forceinline__ void tdm_load_weight(const f16* gsrc, unsigned int lds_byte_off) {
    unsigned long long ga = (unsigned long long)gsrc;
    u32x4 g0;
    g0[0] = 1u;                                         // count=1, user mode
    g0[1] = lds_byte_off;                               // lds_addr
    g0[2] = (unsigned int)ga;                           // global_addr[31:0]
    g0[3] = (unsigned int)((ga >> 32) & 0x01FFFFFFu)    // global_addr[56:32]
          | (2u << 30);                                 // type = 2 ("image")
    i32x8 g1;
    g1[0] = (int)((1u << 16)      // data_size = 2 bytes
                | (1u << 20)      // pad_enable
                | (5u << 22)      // pad_interval: 64 DWORDs (256B) of data...
                | (3u << 25));    // pad_amount:   ...then 4 DWORDs (16B) pad
    g1[1] = (int)(16384u << 16);  // tensor_dim0[15:0] -> [31:16]
    g1[2] = (int)(1u << 16);      // tensor_dim0[31:16]=0 | tensor_dim1=1
    g1[3] = (int)(16384u << 16);  // tensor_dim1 hi=0 | tile_dim0 = 16384
    g1[4] = 1;                    // tile_dim1 = 1, tile_dim2 = 0
    g1[5] = 16384;                // tensor_dim0_stride[31:0]
    g1[6] = 0;
    g1[7] = 0;
    i32x4 z4 = {0, 0, 0, 0};
#if defined(__clang_major__) && (__clang_major__ >= 23)
    i32x8 z8 = {0, 0, 0, 0, 0, 0, 0, 0};
    __builtin_amdgcn_tensor_load_to_lds(g0, g1, z4, z4, z8, 0);
#else
    __builtin_amdgcn_tensor_load_to_lds(g0, g1, z4, z4, 0);
#endif
}

// ---------------- small helpers ----------------
__device__ __forceinline__ void atomicMaxF(float* addr, float val) {
    if (val >= 0.0f) atomicMax((int*)addr, __float_as_int(val));
    else             atomicMin((unsigned int*)addr, __float_as_uint(val));
}

// LayerNorm 128-wide rows into LDS as f16. wave-per-row, 4 floats/lane, shfl reduce.
__device__ __forceinline__ void ln_to_lds(const float* __restrict__ src,
                                          const float* __restrict__ g,
                                          const float* __restrict__ b,
                                          int row0, int rowsInBlock, f16* xs) {
    const int lane = threadIdx.x & 31;
    const int wave = threadIdx.x >> 5;
    for (int r = wave; r < rowsInBlock; r += 8) {
        const float4 xv = *(const float4*)(src + (size_t)(row0 + r) * DD + lane * 4);
        float s  = xv.x + xv.y + xv.z + xv.w;
        float ss = xv.x * xv.x + xv.y * xv.y + xv.z * xv.z + xv.w * xv.w;
#pragma unroll
        for (int m = 16; m >= 1; m >>= 1) {
            s  += __shfl_xor(s,  m, 32);
            ss += __shfl_xor(ss, m, 32);
        }
        const float mean = s * (1.0f / 128.0f);
        const float var  = ss * (1.0f / 128.0f) - mean * mean;
        const float rstd = rsqrtf(var + 1e-5f);
        const float4 gv = *(const float4*)(g + lane * 4);
        const float4 bv = *(const float4*)(b + lane * 4);
        f16* xr = xs + r * XSTR + lane * 4;
        xr[0] = (f16)((xv.x - mean) * rstd * gv.x + bv.x);
        xr[1] = (f16)((xv.y - mean) * rstd * gv.y + bv.y);
        xr[2] = (f16)((xv.z - mean) * rstd * gv.z + bv.z);
        xr[3] = (f16)((xv.w - mean) * rstd * gv.w + bv.w);
    }
}

// ---------------- kernels ----------------

// convert a 128x128 f32 weight to f16, optionally transposed so wT[n][k] = W[k][n]
__global__ void k_cvt_w(const float* __restrict__ W, f16* __restrict__ dst, int transpose) {
    int idx = blockIdx.x * 256 + threadIdx.x;
    if (idx >= 128 * 128) return;
    int kk = idx >> 7, n = idx & 127;
    float v = W[idx];
    if (transpose) dst[n * 128 + kk] = (f16)v;
    else           dst[idx]          = (f16)v;
}

__global__ void k_init(float* __restrict__ amax, float* __restrict__ denom,
                       float* __restrict__ agg) {
    int i = blockIdx.x * 256 + threadIdx.x;
    if (i < NN * HH) { amax[i] = -INFINITY; denom[i] = 0.0f; }
    if (i < NN * DD) agg[i] = 0.0f;
}

// node path: LN(x) -> Q,K,V (f32). Weights DMA'd by TDM while LN runs.
__global__ __launch_bounds__(256) void k_node_qkv(
    const float* __restrict__ x, const float* __restrict__ g, const float* __restrict__ b,
    const f16* __restrict__ WqT, const f16* __restrict__ WkT, const f16* __restrict__ WvT,
    float* __restrict__ q, float* __restrict__ k, float* __restrict__ v) {
    __shared__ f16 smem[4 * TILE_B];          // xs + 3 weight tiles (139,264 B)
    f16* xs = smem;
    const int row0 = blockIdx.x * 128;
    const int rowsInBlock = (NN - row0 < 128) ? (NN - row0) : 128;
    const int lane = threadIdx.x & 31, wave = threadIdx.x >> 5;

    if (wave == 0) {                           // one wave issues the three DMAs
        const unsigned int base = lds_off_u32(smem);
        tdm_load_weight(WqT, base + 1u * TILE_BYTES);
        tdm_load_weight(WkT, base + 2u * TILE_BYTES);
        tdm_load_weight(WvT, base + 3u * TILE_BYTES);
    }
    ln_to_lds(x, g, b, row0, rowsInBlock, xs);   // overlaps with TDM
    if (wave == 0) __builtin_amdgcn_s_wait_tensorcnt((short)0);
    __syncthreads();

    if (wave * 16 >= rowsInBlock) return;      // wave-uniform: EXEC stays all-1s
    const int rbase  = wave * 16;
    const int khalf  = (lane >= 16) ? 8 : 0;
    const int koff   = (lane >= 16) ? 16 : 0;
    const int n15    = lane & 15;
    const int rowadd = (lane >= 16) ? 8 : 0;
    const int arow   = rbase + n15;

    v16h A[4];
#pragma unroll
    for (int kk = 0; kk < 4; ++kk) A[kk] = load_afrag(xs, arow, kk, khalf);

    float* Os[3] = {q, k, v};
#pragma unroll
    for (int m = 0; m < 3; ++m) {
        const f16* wT = smem + (m + 1) * TILE_B;
        float*     O  = Os[m];
#pragma unroll
        for (int t = 0; t < 8; ++t) {
            v8f acc = {};
#pragma unroll
            for (int kk = 0; kk < 4; ++kk)
                acc = wmma_f16(A[kk], load_bfrag(wT, t * 16 + n15, kk * 32 + koff), acc);
            float* op = O + (size_t)(row0 + rbase + rowadd) * DD + t * 16 + n15;
#pragma unroll
            for (int j = 0; j < 8; ++j) op[(size_t)j * DD] = acc[j];
        }
    }
}

// edge path: LN(edges) -> EK, EV (f16)
__global__ __launch_bounds__(256) void k_edge_kv(
    const float* __restrict__ edges, const float* __restrict__ g, const float* __restrict__ b,
    const f16* __restrict__ WekT, const f16* __restrict__ WevT,
    f16* __restrict__ ek, f16* __restrict__ ev) {
    __shared__ f16 smem[3 * TILE_B];          // xs + 2 weight tiles (104,448 B)
    f16* xs = smem;
    const int row0 = blockIdx.x * 128;
    const int rowsInBlock = (EE - row0 < 128) ? (EE - row0) : 128;
    const int lane = threadIdx.x & 31, wave = threadIdx.x >> 5;

    if (wave == 0) {
        const unsigned int base = lds_off_u32(smem);
        tdm_load_weight(WekT, base + 1u * TILE_BYTES);
        tdm_load_weight(WevT, base + 2u * TILE_BYTES);
    }
    ln_to_lds(edges, g, b, row0, rowsInBlock, xs);
    if (wave == 0) __builtin_amdgcn_s_wait_tensorcnt((short)0);
    __syncthreads();

    if (wave * 16 >= rowsInBlock) return;
    const int rbase  = wave * 16;
    const int khalf  = (lane >= 16) ? 8 : 0;
    const int koff   = (lane >= 16) ? 16 : 0;
    const int n15    = lane & 15;
    const int rowadd = (lane >= 16) ? 8 : 0;
    const int arow   = rbase + n15;

    v16h A[4];
#pragma unroll
    for (int kk = 0; kk < 4; ++kk) A[kk] = load_afrag(xs, arow, kk, khalf);

    f16* Os[2] = {ek, ev};
#pragma unroll
    for (int m = 0; m < 2; ++m) {
        const f16* wT = smem + (m + 1) * TILE_B;
        f16*       O  = Os[m];
#pragma unroll
        for (int t = 0; t < 8; ++t) {
            v8f acc = {};
#pragma unroll
            for (int kk = 0; kk < 4; ++kk)
                acc = wmma_f16(A[kk], load_bfrag(wT, t * 16 + n15, kk * 32 + koff), acc);
            f16* op = O + (size_t)(row0 + rbase + rowadd) * DD + t * 16 + n15;
#pragma unroll
            for (int j = 0; j < 8; ++j) op[(size_t)j * DD] = (f16)acc[j];
        }
    }
}

// per-edge: logits -> expand -> exact GELU -> squeeze; atomic segment max on src
__global__ __launch_bounds__(256) void k_logits(
    const float* __restrict__ q, const float* __restrict__ k, const f16* __restrict__ ek,
    const int* __restrict__ ei, const float* __restrict__ Wexp, const float* __restrict__ Wsq,
    float* __restrict__ a_buf, float* __restrict__ amax) {
    __shared__ float sE[64], sS[64];
    if (threadIdx.x < 64) { sE[threadIdx.x] = Wexp[threadIdx.x]; sS[threadIdx.x] = Wsq[threadIdx.x]; }
    __syncthreads();
    const int e = blockIdx.x * 256 + threadIdx.x;
    if (e >= EE) return;
    const int s = ei[e], d = ei[EE + e];
    const float* qr = q + (size_t)s * DD;
    const float* kr = k + (size_t)d * DD;
    const f16*   er = ek + (size_t)e * DD;

    float lg[HH];
#pragma unroll
    for (int hh = 0; hh < HH; ++hh) {
        float acc = 0.0f;
#pragma unroll
        for (int dd = 0; dd < HD; dd += 4) {
            const float4 qv = *(const float4*)(qr + hh * HD + dd);
            const float4 kv = *(const float4*)(kr + hh * HD + dd);
            const v4h    e4 = *(const v4h*)(er + hh * HD + dd);
            acc += qv.x * (kv.x + (float)e4[0]);
            acc += qv.y * (kv.y + (float)e4[1]);
            acc += qv.z * (kv.z + (float)e4[2]);
            acc += qv.w * (kv.w + (float)e4[3]);
        }
        lg[hh] = acc * SCALE;
    }
    float gg[HH];
#pragma unroll
    for (int i = 0; i < HH; ++i) {
        float t = 0.0f;
#pragma unroll
        for (int j = 0; j < HH; ++j) t += lg[j] * sE[i * 8 + j];
        gg[i] = 0.5f * t * (1.0f + erff(t * 0.70710678118f));   // exact GELU
    }
#pragma unroll
    for (int i = 0; i < HH; ++i) {
        float t = 0.0f;
#pragma unroll
        for (int j = 0; j < HH; ++j) t += gg[j] * sS[i * 8 + j];
        a_buf[(size_t)e * HH + i] = t;
        atomicMaxF(amax + (size_t)s * HH + i, t);
    }
}

// ex = exp(a - amax[src]); denominators via f32 atomic add
__global__ __launch_bounds__(256) void k_expsum(
    const int* __restrict__ ei, const float* __restrict__ amax,
    float* __restrict__ a_buf, float* __restrict__ denom) {
    const int e = blockIdx.x * 256 + threadIdx.x;
    if (e >= EE) return;
    const int s = ei[e];
#pragma unroll
    for (int h = 0; h < HH; ++h) {
        const float ex = expf(a_buf[(size_t)e * HH + h] - amax[(size_t)s * HH + h]);
        a_buf[(size_t)e * HH + h] = ex;
        atomicAdd(denom + (size_t)s * HH + h, ex);
    }
}

// attn = ex/denom; write attn.T; scatter attn*(v[dst]+ev) into agg. warp-per-edge.
__global__ __launch_bounds__(256) void k_scatter(
    const float* __restrict__ v, const f16* __restrict__ ev, const int* __restrict__ ei,
    const float* __restrict__ exb, const float* __restrict__ denom,
    float* __restrict__ agg, float* __restrict__ attn_out) {
    const int lane = threadIdx.x & 31;
    const int e = blockIdx.x * 8 + (threadIdx.x >> 5);
    if (e >= EE) return;
    const int s = ei[e], d = ei[EE + e];
    const int h = lane >> 2;
    const float ex  = exb[(size_t)e * HH + h];
    const float den = denom[(size_t)s * HH + h];
    const float at  = ex / (den + 1e-16f);
    if ((lane & 3) == 0) attn_out[(size_t)h * EE + e] = at;   // attn returned as [H,E]
    const int c = lane * 4;
    const float4 vv = *(const float4*)(v + (size_t)d * DD + c);
    const v4h    e4 = *(const v4h*)(ev + (size_t)e * DD + c);
    float* ap = agg + (size_t)s * DD + c;
    atomicAdd(ap + 0, at * (vv.x + (float)e4[0]));
    atomicAdd(ap + 1, at * (vv.y + (float)e4[1]));
    atomicAdd(ap + 2, at * (vv.z + (float)e4[2]));
    atomicAdd(ap + 3, at * (vv.w + (float)e4[3]));
}

// out = agg @ Wo^T + bo   (wT[n][k] = Wo[n][k], direct f16 copy)
__global__ __launch_bounds__(256) void k_out(
    const float* __restrict__ agg, const f16* __restrict__ WoH,
    const float* __restrict__ bo, float* __restrict__ out) {
    __shared__ f16 smem[2 * TILE_B];          // xs + 1 weight tile (69,632 B)
    f16* xs = smem;
    const int row0 = blockIdx.x * 128;
    const int rowsInBlock = (NN - row0 < 128) ? (NN - row0) : 128;
    const int lane = threadIdx.x & 31, wave = threadIdx.x >> 5;

    if (wave == 0)
        tdm_load_weight(WoH, lds_off_u32(smem) + 1u * TILE_BYTES);

    for (int idx = threadIdx.x; idx < 128 * 128; idx += 256) {
        const int r = idx >> 7, c = idx & 127;
        const float val = (r < rowsInBlock) ? agg[(size_t)(row0 + r) * DD + c] : 0.0f;
        xs[r * XSTR + c] = (f16)val;
    }
    if (wave == 0) __builtin_amdgcn_s_wait_tensorcnt((short)0);
    __syncthreads();

    if (wave * 16 >= rowsInBlock) return;
    const int rbase  = wave * 16;
    const int khalf  = (lane >= 16) ? 8 : 0;
    const int koff   = (lane >= 16) ? 16 : 0;
    const int n15    = lane & 15;
    const int rowadd = (lane >= 16) ? 8 : 0;
    const int arow   = rbase + n15;

    v16h A[4];
#pragma unroll
    for (int kk = 0; kk < 4; ++kk) A[kk] = load_afrag(xs, arow, kk, khalf);

    const f16* wT = smem + TILE_B;
#pragma unroll
    for (int t = 0; t < 8; ++t) {
        v8f acc = {};
#pragma unroll
        for (int kk = 0; kk < 4; ++kk)
            acc = wmma_f16(A[kk], load_bfrag(wT, t * 16 + n15, kk * 32 + koff), acc);
        const float bias = bo[t * 16 + n15];
        float* op = out + (size_t)(row0 + rbase + rowadd) * DD + t * 16 + n15;
#pragma unroll
        for (int j = 0; j < 8; ++j) op[(size_t)j * DD] = acc[j] + bias;
    }
}

// ---------------- host launch ----------------
extern "C" void kernel_launch(void* const* d_in, const int* in_sizes, int n_in,
                              void* d_out, int out_size, void* d_ws, size_t ws_size,
                              hipStream_t stream) {
    const float* x     = (const float*)d_in[0];
    const float* edges = (const float*)d_in[1];
    const int*   ei    = (const int*)d_in[2];
    const float* g     = (const float*)d_in[3];
    const float* b     = (const float*)d_in[4];
    const float* Wq    = (const float*)d_in[5];
    const float* Wk    = (const float*)d_in[6];
    const float* Wv    = (const float*)d_in[7];
    const float* Wek   = (const float*)d_in[8];
    const float* Wev   = (const float*)d_in[9];
    const float* Wexp  = (const float*)d_in[10];
    const float* Wsq   = (const float*)d_in[11];
    const float* Wo    = (const float*)d_in[12];
    const float* bo    = (const float*)d_in[13];

    float* out      = (float*)d_out;                    // [N, D]
    float* attn_out = (float*)d_out + (size_t)NN * DD;  // [H, E]

    // workspace carve-up
    char* p = (char*)d_ws;
    float* q     = (float*)p; p += (size_t)NN * DD * 4;
    float* kbuf  = (float*)p; p += (size_t)NN * DD * 4;
    float* vbuf  = (float*)p; p += (size_t)NN * DD * 4;
    float* a_buf = (float*)p; p += (size_t)EE * HH * 4;
    float* amax  = (float*)p; p += (size_t)NN * HH * 4;
    float* denom = (float*)p; p += (size_t)NN * HH * 4;
    float* agg   = (float*)p; p += (size_t)NN * DD * 4;
    f16* WqT  = (f16*)p; p += (size_t)DD * DD * 2;
    f16* WkT  = (f16*)p; p += (size_t)DD * DD * 2;
    f16* WvT  = (f16*)p; p += (size_t)DD * DD * 2;
    f16* WekT = (f16*)p; p += (size_t)DD * DD * 2;
    f16* WevT = (f16*)p; p += (size_t)DD * DD * 2;
    f16* WoH  = (f16*)p; p += (size_t)DD * DD * 2;
    f16* ek   = (f16*)p; p += (size_t)EE * DD * 2;
    f16* ev   = (f16*)p; p += (size_t)EE * DD * 2;

    const int cvtGrid = (128 * 128 + 255) / 256;
    k_cvt_w<<<cvtGrid, 256, 0, stream>>>(Wq,  WqT,  1);
    k_cvt_w<<<cvtGrid, 256, 0, stream>>>(Wk,  WkT,  1);
    k_cvt_w<<<cvtGrid, 256, 0, stream>>>(Wv,  WvT,  1);
    k_cvt_w<<<cvtGrid, 256, 0, stream>>>(Wek, WekT, 1);
    k_cvt_w<<<cvtGrid, 256, 0, stream>>>(Wev, WevT, 1);
    k_cvt_w<<<cvtGrid, 256, 0, stream>>>(Wo,  WoH,  0);

    k_init<<<(NN * DD + 255) / 256, 256, 0, stream>>>(amax, denom, agg);

    k_node_qkv<<<(NN + 127) / 128, 256, 0, stream>>>(x, g, b, WqT, WkT, WvT, q, kbuf, vbuf);
    k_edge_kv <<<(EE + 127) / 128, 256, 0, stream>>>(edges, g, b, WekT, WevT, ek, ev);
    k_logits  <<<(EE + 255) / 256, 256, 0, stream>>>(q, kbuf, ek, ei, Wexp, Wsq, a_buf, amax);
    k_expsum  <<<(EE + 255) / 256, 256, 0, stream>>>(ei, amax, a_buf, denom);
    k_scatter <<<(EE + 7) / 8,     256, 0, stream>>>(vbuf, ev, ei, a_buf, denom, agg, attn_out);
    k_out     <<<(NN + 127) / 128, 256, 0, stream>>>(agg, WoH, bo, out);
}
